// CausalTemporalGNN_49787260895315
// MI455X (gfx1250) — matrix operand order repaired and checked by reading, not compiled, not run on hardware
//
#include <hip/hip_runtime.h>
#include <hip/hip_bf16.h>

#define NUSERS 100000
#define NITEMS 50000
#define NNODES 150000
#define DIM    64
#define TSTEPS 8
#define NHEADS 4
#define HD     16
#define CSTR   0.5f

typedef __attribute__((ext_vector_type(16))) _Float16 v16h;
typedef __attribute__((ext_vector_type(8)))  float    v8f;

// D = A(16x32 f16) x B(32x16 f16) + C(16x16 f32), wave32
__device__ __forceinline__ v8f wmma16(v16h a, v16h b, v8f c) {
    return __builtin_amdgcn_wmma_f32_16x16x32_f16(
        /*neg_a=*/false, a, /*neg_b=*/false, b,
        /*c_mod=*/(short)0, c, /*reuse_a=*/false, /*reuse_b=*/false);
}

// A-matrix 16x32 f16 fragment per ISA 7.12.2:
// lanes 0-15: M=lane,    VGPR j<4 -> K=2j,2j+1 ; VGPR j>=4 -> K=16+2(j-4)..
// lanes 16-31: M=lane-16, same with K offset +8
__device__ __forceinline__ v16h load_a_frag(const float* tile /*[16][64]*/,
                                            int kbase, int lane) {
    const int m = lane & 15, half = lane >> 4;
    v16h a;
#pragma unroll
    for (int j = 0; j < 8; ++j) {
        int kj = (j < 4 ? 2 * j : 16 + 2 * (j - 4)) + 8 * half;
        a[2 * j]     = (_Float16)tile[m * DIM + kbase + kj];
        a[2 * j + 1] = (_Float16)tile[m * DIM + kbase + kj + 1];
    }
    return a;
}

// B-matrix 32x16 f16 fragment (K x N), W row-major [64][64]:
// VGPR j<4, lanes 0-15: rows K=2j,2j+1; VGPR j>=4: K=16+2(j-4)..; lanes 16-31: same with K offset +8, col N=lane%16
__device__ __forceinline__ v16h make_b_frag(const float* __restrict__ W,
                                            int kbase, int ctile, int lane) {
    const int n = (lane & 15) + ctile * 16, half = lane >> 4;
    v16h b;
#pragma unroll
    for (int j = 0; j < 8; ++j) {
        int k0 = kbase + (j < 4 ? 2 * j : 16 + 2 * (j - 4)) + 8 * half;
        b[2 * j]     = (_Float16)W[k0 * DIM + n];
        b[2 * j + 1] = (_Float16)W[(k0 + 1) * DIM + n];
    }
    return b;
}

// packed layout: frag[((mat*2 + kc)*4 + ct)*32 + lane], mats: q,k,v,o,ws0,wn0,ws1,wn1
#define MAT_STRIDE 256  // v16h entries per matrix (2 kc * 4 ct * 32 lanes)

__global__ void pack_weights_kernel(const float* __restrict__ W0, const float* __restrict__ W1,
                                    const float* __restrict__ W2, const float* __restrict__ W3,
                                    const float* __restrict__ W4, const float* __restrict__ W5,
                                    const float* __restrict__ W6, const float* __restrict__ W7,
                                    v16h* __restrict__ packed) {
    const float* Ws[8] = {W0, W1, W2, W3, W4, W5, W6, W7};
    for (int idx = threadIdx.x; idx < 8 * 2 * 4 * 32; idx += blockDim.x) {
        int lane = idx & 31;
        int ct   = (idx >> 5) & 3;
        int kc   = (idx >> 7) & 1;
        int mat  = idx >> 8;
        packed[idx] = make_b_frag(Ws[mat], kc * 32, ct, lane);
    }
}

// ---------------- temporal projection tables: tq/tk/tv[t][d] = te[t]@W + b ----
__global__ void precompute_tqkv_kernel(const float* __restrict__ te,
                                       const float* __restrict__ Wq, const float* __restrict__ bq,
                                       const float* __restrict__ Wk, const float* __restrict__ bk,
                                       const float* __restrict__ Wv, const float* __restrict__ bv,
                                       float* __restrict__ tqkv /* [3][T][64] */) {
    const int d = threadIdx.x;  // 64 threads
    for (int t = 0; t < TSTEPS; ++t) {
        float aq = bq[d], ak = bk[d], av = bv[d];
        for (int k = 0; k < DIM; ++k) {
            float e = te[t * DIM + k];
            aq += e * Wq[k * DIM + d];
            ak += e * Wk[k * DIM + d];
            av += e * Wv[k * DIM + d];
        }
        tqkv[(0 * TSTEPS + t) * DIM + d] = aq;
        tqkv[(1 * TSTEPS + t) * DIM + d] = ak;
        tqkv[(2 * TSTEPS + t) * DIM + d] = av;
    }
}

// ---------------- fused temporal MHA (+ residual) --------------------------
// block = 64 threads (2 waves), each wave owns a 16-node tile.
__global__ void __launch_bounds__(64)
temporal_attn_kernel(const float* __restrict__ user_emb, const float* __restrict__ item_emb,
                     const float* __restrict__ temporal_emb, const int* __restrict__ time_indices,
                     const v16h* __restrict__ wpack, const float* __restrict__ bo,
                     const float* __restrict__ tqkv, float* __restrict__ x_out) {
    __shared__ float xs[2][16][DIM];
    __shared__ float qs[2][16][DIM];
    __shared__ float ksh[2][16][DIM];
    __shared__ float vsh[2][16][DIM];
    __shared__ float osh[2][16][DIM];
    __shared__ float tqs[3 * TSTEPS * DIM];
    __shared__ int   tsh[2][16];

    const int w = threadIdx.x >> 5;
    const int lane = threadIdx.x & 31;
    const int tileBase = blockIdx.x * 32 + w * 16;

    // stage temporal tables (block-wide)
    for (int i = threadIdx.x; i < 3 * TSTEPS * DIM; i += 64) tqs[i] = tqkv[i];

    // stage x tile (coalesced), zero-pad the tail
#pragma unroll 4
    for (int r = 0; r < 16; ++r) {
        int node = tileBase + r;
        float v0 = 0.f, v1 = 0.f;
        if (node < NNODES) {
            const float* row = (node < NUSERS) ? user_emb + (size_t)node * DIM
                                               : item_emb + (size_t)(node - NUSERS) * DIM;
            v0 = row[lane];
            v1 = row[lane + 32];
        }
        xs[w][r][lane] = v0;
        xs[w][r][lane + 32] = v1;
    }
    if (lane < 16) {
        int node = tileBase + lane;
        tsh[w][lane] = (node < NNODES) ? time_indices[node] : 0;
    }

    // A fragments of the x tile (same wave wrote xs -> in-order, no barrier needed)
    v16h a0 = load_a_frag(&xs[w][0][0], 0, lane);
    v16h a1 = load_a_frag(&xs[w][0][0], 32, lane);

    const int nloc = lane & 15, half = lane >> 4;
    float (*dst[3])[DIM] = {qs[w], ksh[w], vsh[w]};
#pragma unroll
    for (int m = 0; m < 3; ++m) {
        const v16h* wf = wpack + m * MAT_STRIDE;
#pragma unroll
        for (int ct = 0; ct < 4; ++ct) {
            v8f acc = {};
            acc = wmma16(a0, wf[(0 * 4 + ct) * 32 + lane], acc);
            acc = wmma16(a1, wf[(1 * 4 + ct) * 32 + lane], acc);
            int n = nloc + ct * 16;
#pragma unroll
            for (int r = 0; r < 8; ++r) dst[m][r + 8 * half][n] = acc[r];
        }
    }
    __syncthreads();

    // scalar attention: only the query at t = time_indices[node] matters.
    // 2 tiles * 16 nodes * 4 heads = 128 tasks on 64 threads.
#pragma unroll
    for (int it = 0; it < 2; ++it) {
        int task = threadIdx.x + 64 * it;
        int nl = task >> 2, h = task & 3;
        int tile = nl >> 4, r = nl & 15;
        int tq_i = tsh[tile][r];
        const float* tq = tqs + (0 * TSTEPS + tq_i) * DIM + h * HD;
        float qh[HD];
#pragma unroll
        for (int i = 0; i < HD; ++i) qh[i] = qs[tile][r][h * HD + i] + tq[i];
        float sc[TSTEPS], mx = -1e30f;
#pragma unroll
        for (int t = 0; t < TSTEPS; ++t) {
            const float* tk = tqs + (1 * TSTEPS + t) * DIM + h * HD;
            float s = 0.f;
#pragma unroll
            for (int i = 0; i < HD; ++i) s += qh[i] * (ksh[tile][r][h * HD + i] + tk[i]);
            s *= 0.25f;  // 1/sqrt(16)
            sc[t] = s;
            mx = fmaxf(mx, s);
        }
        float se = 0.f;
#pragma unroll
        for (int t = 0; t < TSTEPS; ++t) { sc[t] = __expf(sc[t] - mx); se += sc[t]; }
        float inv = 1.f / se;
        float out[HD];
#pragma unroll
        for (int i = 0; i < HD; ++i) out[i] = 0.f;
#pragma unroll
        for (int t = 0; t < TSTEPS; ++t) {
            const float* tv = tqs + (2 * TSTEPS + t) * DIM + h * HD;
            float p = sc[t] * inv;
#pragma unroll
            for (int i = 0; i < HD; ++i) out[i] += p * (vsh[tile][r][h * HD + i] + tv[i]);
        }
#pragma unroll
        for (int i = 0; i < HD; ++i) osh[tile][r][h * HD + i] = out[i];
    }
    __syncthreads();

    // attended @ Wo + bo, then residual x + att + te[t]
    v16h o0 = load_a_frag(&osh[w][0][0], 0, lane);
    v16h o1 = load_a_frag(&osh[w][0][0], 32, lane);
    const v16h* wo = wpack + 3 * MAT_STRIDE;
#pragma unroll
    for (int ct = 0; ct < 4; ++ct) {
        v8f acc = {};
        acc = wmma16(o0, wo[(0 * 4 + ct) * 32 + lane], acc);
        acc = wmma16(o1, wo[(1 * 4 + ct) * 32 + lane], acc);
        int n = nloc + ct * 16;
#pragma unroll
        for (int r = 0; r < 8; ++r) {
            int rr = r + 8 * half;
            int node = tileBase + rr;
            if (node < NNODES) {
                int ti = tsh[w][rr];
                float val = acc[r] + bo[n] + xs[w][rr][n] + temporal_emb[ti * DIM + n];
                x_out[(size_t)node * DIM + n] = val;
            }
        }
    }
}

// ---------------- h = x + 0.5*causal ; agg = 0 ------------------------------
__global__ void prep_kernel(const float* __restrict__ x, const float* __restrict__ causal,
                            float* __restrict__ h, float* __restrict__ agg) {
    size_t i = (size_t)blockIdx.x * blockDim.x + threadIdx.x;
    if (i < (size_t)NNODES * DIM) {
        h[i] = x[i] + CSTR * causal[i];
        agg[i] = 0.f;
    }
}

__global__ void zero_kernel(float* __restrict__ agg) {
    size_t i = (size_t)blockIdx.x * blockDim.x + threadIdx.x;
    if (i < (size_t)NNODES * DIM) agg[i] = 0.f;
}

// ---------------- edge scatter: agg[dst] += h[src] (L2-resident atomics) ----
__global__ void scatter_kernel(const int* __restrict__ src, const int* __restrict__ dst,
                               const float* __restrict__ h, float* __restrict__ agg,
                               long long nedges) {
    const long long total = nedges * DIM;
    const long long stride = (long long)gridDim.x * blockDim.x;
    for (long long gid = (long long)blockIdx.x * blockDim.x + threadIdx.x;
         gid < total; gid += stride) {
        long long e = gid >> 6;
        int d = (int)(gid & 63);
        // prefetch the next iteration's edge-index chunk (global_prefetch_b8)
        long long en = (gid + stride) >> 6;
        if (d == 0 && en < nedges) {
            __builtin_prefetch(&src[en], 0, 0);
            __builtin_prefetch(&dst[en], 0, 0);
        }
        int s = src[e], t = dst[e];
        atomicAdd(&agg[(size_t)t * DIM + d], h[(size_t)s * DIM + d]);
    }
}

// ---------------- out = relu(h@Wself + agg@Wneigh + b) ----------------------
__global__ void __launch_bounds__(128)
graph_combine_kernel(const float* __restrict__ h, const float* __restrict__ agg,
                     const v16h* __restrict__ wself_f, const v16h* __restrict__ wneigh_f,
                     const float* __restrict__ bias, float* __restrict__ out) {
    __shared__ float hs[4][16][DIM];
    __shared__ float gs[4][16][DIM];
    const int w = threadIdx.x >> 5;
    const int lane = threadIdx.x & 31;
    const int tileBase = blockIdx.x * 64 + w * 16;

#pragma unroll 4
    for (int r = 0; r < 16; ++r) {
        int node = tileBase + r;
        float h0 = 0.f, h1 = 0.f, g0 = 0.f, g1 = 0.f;
        if (node < NNODES) {
            size_t o = (size_t)node * DIM;
            h0 = h[o + lane]; h1 = h[o + lane + 32];
            g0 = agg[o + lane]; g1 = agg[o + lane + 32];
        }
        hs[w][r][lane] = h0; hs[w][r][lane + 32] = h1;
        gs[w][r][lane] = g0; gs[w][r][lane + 32] = g1;
    }

    v16h ha0 = load_a_frag(&hs[w][0][0], 0, lane);
    v16h ha1 = load_a_frag(&hs[w][0][0], 32, lane);
    v16h ga0 = load_a_frag(&gs[w][0][0], 0, lane);
    v16h ga1 = load_a_frag(&gs[w][0][0], 32, lane);

    const int nloc = lane & 15, half = lane >> 4;
#pragma unroll
    for (int ct = 0; ct < 4; ++ct) {
        v8f acc = {};
        acc = wmma16(ha0, wself_f[(0 * 4 + ct) * 32 + lane], acc);
        acc = wmma16(ha1, wself_f[(1 * 4 + ct) * 32 + lane], acc);
        acc = wmma16(ga0, wneigh_f[(0 * 4 + ct) * 32 + lane], acc);
        acc = wmma16(ga1, wneigh_f[(1 * 4 + ct) * 32 + lane], acc);
        int n = nloc + ct * 16;
#pragma unroll
        for (int r = 0; r < 8; ++r) {
            int node = tileBase + r + 8 * half;
            if (node < NNODES) {
                float v = acc[r] + bias[n];
                out[(size_t)node * DIM + n] = v > 0.f ? v : 0.f;
            }
        }
    }
}

// ---------------- final: all = x + h ; user/item views are the same split ---
__global__ void finalize_kernel(const float* __restrict__ x, const float* __restrict__ h,
                                float* __restrict__ out) {
    size_t i = (size_t)blockIdx.x * blockDim.x + threadIdx.x;
    const size_t total = (size_t)NNODES * DIM;
    if (i < total) {
        float v = x[i] + h[i];
        out[i] = v;            // all_embeddings
        out[total + i] = v;    // user section then item section — contiguous split
    }
}

extern "C" void kernel_launch(void* const* d_in, const int* in_sizes, int n_in,
                              void* d_out, int out_size, void* d_ws, size_t ws_size,
                              hipStream_t stream) {
    const int*   edge_index   = (const int*)d_in[0];
    const int*   time_indices = (const int*)d_in[2];
    const float* user_emb     = (const float*)d_in[3];
    const float* item_emb     = (const float*)d_in[4];
    const float* temporal_emb = (const float*)d_in[5];
    const float* causal_emb   = (const float*)d_in[6];
    const float* Wq = (const float*)d_in[7];
    const float* bq = (const float*)d_in[8];
    const float* Wk = (const float*)d_in[9];
    const float* bk = (const float*)d_in[10];
    const float* Wv = (const float*)d_in[11];
    const float* bv = (const float*)d_in[12];
    const float* Wo = (const float*)d_in[13];
    const float* bo = (const float*)d_in[14];
    const float* Ws0 = (const float*)d_in[15];
    const float* Wn0 = (const float*)d_in[16];
    const float* b0  = (const float*)d_in[17];
    const float* Ws1 = (const float*)d_in[18];
    const float* Wn1 = (const float*)d_in[19];
    const float* b1  = (const float*)d_in[20];

    const long long nedges = (long long)in_sizes[0] / 2;
    const size_t nd = (size_t)NNODES * DIM;

    float* buf_x   = (float*)d_ws;
    float* buf_h   = buf_x + nd;
    float* buf_agg = buf_h + nd;
    float* buf_h2  = buf_agg + nd;
    float* tqkv    = buf_h2 + nd;                  // [3][T][64]
    v16h*  wpack   = (v16h*)(tqkv + 3 * TSTEPS * DIM);  // 8 * 256 v16h (64 KB), 32B aligned

    const int ew = 256;
    const unsigned eb_nodes = (unsigned)((nd + ew - 1) / ew);
    const unsigned eb_edges = 32768;  // grid-stride scatter

    // 0) pack the 8 weight matrices into f16 per-lane B-fragments (once per call)
    pack_weights_kernel<<<1, 256, 0, stream>>>(Wq, Wk, Wv, Wo, Ws0, Wn0, Ws1, Wn1, wpack);

    // 1) temporal projection tables
    precompute_tqkv_kernel<<<1, 64, 0, stream>>>(temporal_emb, Wq, bq, Wk, bk, Wv, bv, tqkv);

    // 2) fused temporal MHA + residual -> buf_x  (WMMA GEMMs, scalar softmax)
    temporal_attn_kernel<<<(NNODES + 31) / 32, 64, 0, stream>>>(
        user_emb, item_emb, temporal_emb, time_indices, wpack, bo, tqkv, buf_x);

    // 3) h0 = x + 0.5*causal ; zero agg
    prep_kernel<<<eb_nodes, ew, 0, stream>>>(buf_x, causal_emb, buf_h, buf_agg);

    // 4) layer 0 scatter + combine
    scatter_kernel<<<eb_edges, ew, 0, stream>>>(edge_index, edge_index + nedges,
                                                buf_h, buf_agg, nedges);
    graph_combine_kernel<<<(NNODES + 63) / 64, 128, 0, stream>>>(
        buf_h, buf_agg, wpack + 4 * MAT_STRIDE, wpack + 5 * MAT_STRIDE, b0, buf_h2);

    // 5) layer 1 scatter + combine
    zero_kernel<<<eb_nodes, ew, 0, stream>>>(buf_agg);
    scatter_kernel<<<eb_edges, ew, 0, stream>>>(edge_index, edge_index + nedges,
                                                buf_h2, buf_agg, nedges);
    graph_combine_kernel<<<(NNODES + 63) / 64, 128, 0, stream>>>(
        buf_h2, buf_agg, wpack + 6 * MAT_STRIDE, wpack + 7 * MAT_STRIDE, b1, buf_h);

    // 6) all = x + h ; user/item sections
    finalize_kernel<<<eb_nodes, ew, 0, stream>>>(buf_x, buf_h, (float*)d_out);
}
